// MSDA3D_33715493274322
// MI455X (gfx1250) — matrix operand order repaired and checked by reading, not compiled, not run on hardware
//
#include <hip/hip_runtime.h>
#include <hip/hip_bf16.h>
#include <math.h>

// ---------------------------------------------------------------------------
// MSDA3D for MI455X (gfx1250, wave32).
//   - One-pass f32->f16 conversion (v_cvt_pk_rtz_f16_f32) so GEMM hot loops
//     are pure global_load_b128 + v_wmma_f32_16x16x32_f16.
//   - GEMM: wave owns NT 16-wide N tiles; K loop is explicitly double-
//     buffered (prefetch k+1 fragments before k's WMMA chain) so WMMAs wait
//     on loads issued a full K-step earlier.
//   - val stored f16: halves the dominant L2 gather traffic of the trilinear
//     sampling step (~1.4 GB -> ~0.7 GB).
//   - Sampling: one wave per (query, head); lane = channel (D=32 == wave32),
//     coalesced corner gathers, wave-uniform control flow.
// ---------------------------------------------------------------------------

typedef __attribute__((ext_vector_type(2)))  _Float16 v2h;
typedef __attribute__((ext_vector_type(8)))  _Float16 v8h;
typedef __attribute__((ext_vector_type(16))) _Float16 v16h;
typedef __attribute__((ext_vector_type(8)))  float    v8f;

#define MSDA_H    8
#define MSDA_P    4
#define MSDA_D    32
#define MSDA_DIN  256
#define MSDA_L    4

union Frag16 {
  v16h v;
  struct { v8h lo, hi; } s;
};

// ---- packed f32 -> f16 conversion (2 elements / thread / v_cvt_pk_rtz) ----
__global__ __launch_bounds__(256)
void msda_cvt_f32_f16(const float* __restrict__ in, _Float16* __restrict__ out,
                      long n2) {
  const long i = (long)blockIdx.x * blockDim.x + threadIdx.x;
  if (i < n2) {
    float2 f = ((const float2*)in)[i];
    ((v2h*)out)[i] = __builtin_bit_cast(v2h, __builtin_amdgcn_cvt_pkrtz(f.x, f.y));
  }
}

// ---- C[M,N] = A[M,K=256] @ W[N,K]^T + bias[N]; f16 WMMA, OutT output ------
// block = 128 threads (4 waves, 4 consecutive M tiles); wave covers NT
// consecutive 16-wide N tiles (grid.y = N / (16*NT)).
template <int NT, typename OutT>
__global__ __launch_bounds__(128)
void msda_gemm_f16_wmma(const _Float16* __restrict__ A,
                        const _Float16* __restrict__ W,
                        const float* __restrict__ bias,
                        OutT* __restrict__ C,
                        int M, int N) {
  constexpr int K = MSDA_DIN;
  const int lane = threadIdx.x & 31;
  const int wave = threadIdx.x >> 5;
  const int half = lane >> 4;      // 0: lanes 0-15, 1: lanes 16-31
  const int r    = lane & 15;

  const int tileM   = blockIdx.x * 4 + wave;
  const int colBase = blockIdx.y * (NT * 16);
  if (tileM * 16 >= M) return;

  // Hoisted row base pointers: K-loop loads are pure immediate-offset.
  const _Float16* __restrict__ arow = A + (size_t)(tileM * 16 + r) * K;
  const _Float16* wbase[NT];
#pragma unroll
  for (int j = 0; j < NT; ++j)
    wbase[j] = W + (size_t)(colBase + j * 16 + r) * K + half * 16;

  // A fragment 16x32: lane(0-15)=row; elems 0..15 <- K k0+half*16+0..15.
  // B fragment 32x16: B[k][n] = W[col][k0+k];
  // lanes 0-15 K=0..15, lanes 16-31 K=16..31.  2 x b128 each.
  auto loadA = [&](int k0, Frag16& a) {
    a.s.lo = *(const v8h*)(arow + k0 + half * 16);
    a.s.hi = *(const v8h*)(arow + k0 + half * 16 + 8);
  };
  auto loadB = [&](int k0, Frag16* b) {
#pragma unroll
    for (int j = 0; j < NT; ++j) {
      b[j].s.lo = *(const v8h*)(wbase[j] + k0);
      b[j].s.hi = *(const v8h*)(wbase[j] + k0 + 8);
    }
  };

  // Software pipeline: fragments for step k are loaded one K-step ahead.
  Frag16 aCur, bCur[NT];
  loadA(0, aCur);
  loadB(0, bCur);

  v8f acc[NT] = {};
#pragma unroll
  for (int k0 = 0; k0 < K; k0 += 32) {
    Frag16 aNxt, bNxt[NT];
    if (k0 + 32 < K) {            // compile-time per unrolled iteration
      loadA(k0 + 32, aNxt);
      loadB(k0 + 32, bNxt);
    }
#pragma unroll
    for (int j = 0; j < NT; ++j) {
      acc[j] = __builtin_amdgcn_wmma_f32_16x16x32_f16(
          /*neg_a=*/false, aCur.v, /*neg_b=*/false, bCur[j].v,
          /*c_mod=*/(short)0, acc[j], /*reuse_a=*/false, /*reuse_b=*/false);
    }
    aCur = aNxt;                  // pure SSA renaming after full unroll
#pragma unroll
    for (int j = 0; j < NT; ++j) bCur[j] = bNxt[j];
  }

#pragma unroll
  for (int j = 0; j < NT; ++j) {
    const int   col  = colBase + j * 16 + r;
    const float bcol = bias[col];
#pragma unroll
    for (int v = 0; v < 8; ++v) {
      const int rowOut = tileM * 16 + half * 8 + v;  // C: M = vgpr + 8*half
      C[(size_t)rowOut * N + col] = (OutT)(acc[j][v] + bcol);
    }
  }
}

// ---- softmax + tanh + trilinear sampling; val in f16, out in f16 ----------
__global__ __launch_bounds__(256)
void msda_sample_kernel(const float*    __restrict__ priors,    // [B,Nq,2]
                        const int*      __restrict__ map_hw,    // [4,2] (H,W)
                        const int*      __restrict__ map_offs,  // [4]
                        const int*      __restrict__ map_ids,   // [B,Nq]
                        const float*    __restrict__ offs_raw,  // [B*Nq,96]
                        const float*    __restrict__ attn_raw,  // [B*Nq,32]
                        const _Float16* __restrict__ val,       // [B,Ns,256] f16
                        _Float16*       __restrict__ attn_feats,// [B*Nq,256] f16
                        int Nq, int Ns, long n_tasks) {
  const int  lane = threadIdx.x & 31;
  const int  wave = threadIdx.x >> 5;
  const long task = (long)blockIdx.x * (blockDim.x >> 5) + wave;
  if (task >= n_tasks) return;

  const int  h  = (int)(task & (MSDA_H - 1));
  const long bq = task >> 3;                 // b*Nq + q
  const int  b  = (int)(bq / Nq);

  const float px  = priors[bq * 2 + 0];
  const float py  = priors[bq * 2 + 1];
  const int   mid = map_ids[bq];
  const float normw = (float)map_hw[mid * 2 + 1];   // norm = (W,H)
  const float normh = (float)map_hw[mid * 2 + 0];

  // softmax over P=4 points of this head (wave-uniform scalars)
  float araw[MSDA_P];
  float amax = -3.402823e38f;
#pragma unroll
  for (int p = 0; p < MSDA_P; ++p) {
    araw[p] = attn_raw[bq * 32 + h * MSDA_P + p];
    amax = fmaxf(amax, araw[p]);
  }
  float asum = 0.f;
#pragma unroll
  for (int p = 0; p < MSDA_P; ++p) {
    araw[p] = __expf(araw[p] - amax);
    asum += araw[p];
  }
  const float ainv = 1.f / asum;

  const _Float16* __restrict__ vbase =
      val + (size_t)b * Ns * MSDA_DIN + h * MSDA_D + lane;

  float acc = 0.f;
#pragma unroll
  for (int p = 0; p < MSDA_P; ++p) {
    const float* o = offs_raw + bq * 96 + (h * MSDA_P + p) * 3;
    const float ox = o[0], oy = o[1], oz = o[2];
    const float sx = px + ox / normw;
    const float sy = py + oy / normh;

    const float lvl_f = (float)mid + tanhf(oz);     // z * (L-1)
    const int   l0 = (int)floorf(lvl_f);
    const float wz = lvl_f - (float)l0;

    float s = 0.f;
#pragma unroll
    for (int li = 0; li < 2; ++li) {
      const int   lvl    = l0 + li;
      const float wl     = li ? wz : (1.f - wz);
      const bool  validl = (lvl >= 0) && (lvl < MSDA_L);
      const int   lc     = min(max(lvl, 0), MSDA_L - 1);
      const int   Hl     = map_hw[lc * 2 + 0];
      const int   Wl     = map_hw[lc * 2 + 1];
      const int   off    = map_offs[lc];

      const float xf = sx * (float)Wl - 0.5f;
      const float yf = sy * (float)Hl - 0.5f;
      const int   x0 = (int)floorf(xf);
      const int   y0 = (int)floorf(yf);
      const float dx = xf - (float)x0;
      const float dy = yf - (float)y0;

      float lv = 0.f;
#pragma unroll
      for (int c = 0; c < 4; ++c) {               // bilinear corners
        const int   xi = x0 + (c & 1);
        const int   yi = y0 + (c >> 1);
        const float w  = ((c & 1) ? dx : 1.f - dx) * ((c >> 1) ? dy : 1.f - dy);
        const bool  ok = validl && (xi >= 0) && (xi < Wl) && (yi >= 0) && (yi < Hl);
        if (ok) {                                 // wave-uniform branch
          const long idx = (long)off + (long)yi * Wl + xi;
          lv += w * (float)vbase[(size_t)idx * MSDA_DIN]; // coalesced 64B gather
        }
      }
      s += wl * lv;
    }
    acc += araw[p] * ainv * s;
  }

  attn_feats[bq * MSDA_DIN + h * MSDA_D + lane] = (_Float16)acc;
}

extern "C" void kernel_launch(void* const* d_in, const int* in_sizes, int n_in,
                              void* d_out, int out_size, void* d_ws, size_t ws_size,
                              hipStream_t stream) {
  const float* in_feats = (const float*)d_in[0];
  const float* priors   = (const float*)d_in[1];
  const float* sfeats   = (const float*)d_in[2];
  const int*   map_hw   = (const int*)d_in[3];
  const int*   map_offs = (const int*)d_in[4];
  const int*   map_ids  = (const int*)d_in[5];
  const float* Wo       = (const float*)d_in[6];
  const float* bo       = (const float*)d_in[7];
  const float* Wa       = (const float*)d_in[8];
  const float* ba       = (const float*)d_in[9];
  const float* Wv       = (const float*)d_in[10];
  const float* bv       = (const float*)d_in[11];
  const float* Wout     = (const float*)d_in[12];
  const float* bout     = (const float*)d_in[13];

  const int B  = 2;
  const int Nq = in_sizes[1] / (B * 2);          // sample_priors [B,Nq,2]
  const int Ns = in_sizes[2] / (B * MSDA_DIN);   // sample_feats  [B,Ns,256]
  const int MQ = B * Nq;
  const int MS = B * Ns;

  // ---- workspace partition (all region sizes are multiples of 256 B) ----
  char* ws = (char*)d_ws;
  _Float16* sf16   = (_Float16*)ws;                ws += (size_t)MS * MSDA_DIN * 2;
  _Float16* if16   = (_Float16*)ws;                ws += (size_t)MQ * MSDA_DIN * 2;
  _Float16* Wv16   = (_Float16*)ws;                ws += (size_t)MSDA_DIN * MSDA_DIN * 2;
  _Float16* Wo16   = (_Float16*)ws;                ws += (size_t)96 * MSDA_DIN * 2;
  _Float16* Wa16   = (_Float16*)ws;                ws += (size_t)32 * MSDA_DIN * 2;
  _Float16* Wout16 = (_Float16*)ws;                ws += (size_t)MSDA_DIN * MSDA_DIN * 2;
  _Float16* val16  = (_Float16*)ws;                ws += (size_t)MS * MSDA_DIN * 2;
  float*    offs_rawp  = (float*)ws;               ws += (size_t)MQ * 96 * 4;
  float*    attn_rawp  = (float*)ws;               ws += (size_t)MQ * 32 * 4;
  _Float16* af16   = (_Float16*)ws;                ws += (size_t)MQ * MSDA_DIN * 2;

  // ---- 0) one-pass f32 -> f16 conversions (packed cvt) ----
  auto cvt = [&](const float* src, _Float16* dst, long n) {
    const long n2 = n / 2;
    msda_cvt_f32_f16<<<dim3((unsigned)((n2 + 255) / 256)), dim3(256), 0, stream>>>(
        src, dst, n2);
  };
  cvt(sfeats,   sf16,   (long)MS * MSDA_DIN);
  cvt(in_feats, if16,   (long)MQ * MSDA_DIN);
  cvt(Wv,       Wv16,   (long)MSDA_DIN * MSDA_DIN);
  cvt(Wo,       Wo16,   (long)96 * MSDA_DIN);
  cvt(Wa,       Wa16,   (long)32 * MSDA_DIN);
  cvt(Wout,     Wout16, (long)MSDA_DIN * MSDA_DIN);

  const dim3 blk(128);

  // ---- 1) val16 = sample_feats @ Wv^T + bv  (N=256, NT=4, f16 out) ----
  msda_gemm_f16_wmma<4, _Float16>
      <<<dim3((MS + 63) / 64, MSDA_DIN / 64), blk, 0, stream>>>(
          sf16, Wv16, bv, val16, MS, MSDA_DIN);

  // ---- 2) offs_raw = in_feats @ Wo^T + bo  (N=96, NT=3 x grid.y=2) ----
  msda_gemm_f16_wmma<3, float>
      <<<dim3((MQ + 63) / 64, 2), blk, 0, stream>>>(
          if16, Wo16, bo, offs_rawp, MQ, 96);

  // ---- 3) attn_raw = in_feats @ Wa^T + ba  (N=32, NT=2, f32 out) ----
  msda_gemm_f16_wmma<2, float>
      <<<dim3((MQ + 63) / 64, 1), blk, 0, stream>>>(
          if16, Wa16, ba, attn_rawp, MQ, 32);

  // ---- 4) softmax + tanh + trilinear sampling -> af16 ----
  const long n_tasks = (long)MQ * MSDA_H;
  const int  wpb     = 256 / 32;
  msda_sample_kernel<<<dim3((unsigned)((n_tasks + wpb - 1) / wpb)), dim3(256), 0, stream>>>(
      priors, map_hw, map_offs, map_ids, offs_rawp, attn_rawp, val16, af16,
      Nq, Ns, n_tasks);

  // ---- 5) out = attn_feats @ Wout^T + bout  (N=256, NT=4, f32 out) ----
  msda_gemm_f16_wmma<4, float>
      <<<dim3((MQ + 63) / 64, MSDA_DIN / 64), blk, 0, stream>>>(
          af16, Wout16, bout, (float*)d_out, MQ, MSDA_DIN);
}
